// HGTDetector_87591563034835
// MI455X (gfx1250) — matrix operand order repaired and checked by reading, not compiled, not run on hardware
//
#include <hip/hip_runtime.h>
#include <math.h>

// ---------------------------------------------------------------------------
// CDNA5 (gfx1250) implementation of the HGT bot-detector forward pass.
// All 128-wide GEMMs go through v_wmma_f32_16x16x32_bf16 (fp32->bf16 staged
// in LDS, fp32 accumulation). Edge attention uses wave-per-edge kernels.
// ---------------------------------------------------------------------------

typedef __bf16 bf16_t;
typedef __attribute__((ext_vector_type(16))) __bf16 v16bf;
typedef __attribute__((ext_vector_type(8)))  __bf16 v8bf;
typedef __attribute__((ext_vector_type(4)))  __bf16 v4bf;
typedef __attribute__((ext_vector_type(8)))  float  v8f;

#define LDS_PAD 40   // 32 bf16 payload + 8 pad -> 80B row pitch (mult of 16B)

__device__ __forceinline__ float leaky01(float x) { return x > 0.f ? x : 0.01f * x; }
__device__ __forceinline__ float gelu_exact(float x) {
  return 0.5f * x * (1.f + erff(x * 0.7071067811865475f));
}

// ---------------------------------------------------------------------------
// fill kernel (d_ws is poisoned; we must init agg / m / z ourselves)
// ---------------------------------------------------------------------------
__global__ void fill_f32(float* __restrict__ p, size_t n, float v) {
  size_t i = (size_t)blockIdx.x * blockDim.x + threadIdx.x;
  if (i < n) p[i] = v;
}

// ---------------------------------------------------------------------------
// Generic tiled GEMM: C[M,128] = epilogue( act_in(A[M,K]) @ B[K,128] + bias )
//   gelu_in : apply exact GELU to A elements while staging (node_out path)
//   act_out : 1 -> leaky_relu(0.01)
//   resid   : if non-null, C = sigmoid(*skip)*C + (1-sigmoid(*skip))*resid
// K must be a multiple of 32 (128 / 768 here). One block = 128x128 tile,
// 8 waves, each wave computes a 16x128 strip with 8 WMMA accumulators.
// ---------------------------------------------------------------------------
__global__ __launch_bounds__(256)
void gemm128_wmma_bf16(const float* __restrict__ A, int M, int K,
                       const float* __restrict__ B,     // [K,128] row-major
                       const float* __restrict__ bias,  // [128] or null
                       float* __restrict__ C,           // [M,128]
                       int act_out, int gelu_in,
                       const float* __restrict__ resid, // [M,128] or null
                       const float* __restrict__ skip)  // scalar or null
{
  __shared__ bf16_t As[128][LDS_PAD];   // A tile, row-major [m][k]
  __shared__ bf16_t Bt[128][LDS_PAD];   // B tile transposed [n][k]

  const int tid   = threadIdx.x;
  const int lane  = tid & 31;
  const int wv    = tid >> 5;            // wave 0..7
  const int mBase = blockIdx.x * 128;

  v8f zero = {};
  v8f acc[8];
#pragma unroll
  for (int i = 0; i < 8; ++i) acc[i] = zero;

  // A staging: thread -> 1 row x 16 cols (two b128 LDS stores)
  const int aRow = tid >> 1;             // 0..127
  const int aCol = (tid & 1) * 16;       // 0 or 16
  // B staging: thread -> 4 k-rows x 4 n-cols micro-block (register transpose,
  // four 8B LDS stores; global side stays fully coalesced per k-row)
  const int bK0  = (tid >> 5) * 4;       // 0,4,..,28
  const int bN0  = (tid & 31) * 4;       // 0,4,..,124

  // fragment geometry (ISA 16-bit WMMA layouts, wave32)
  const int mr   = wv * 16 + (lane & 15);
  const int kh   = (lane >> 4) * 8;      // A: lanes 0-15 K{0..7,16..23}; 16-31 K{8..15,24..31}
  const int colb = lane & 15;
  const int kh2  = (lane >> 4) * 8;      // B: lanes 0-15 K{0..7,16..23}; 16-31 K{8..15,24..31}

  for (int kb = 0; kb < K; kb += 32) {
    // ---- prefetch next K-tile (global_prefetch_b8) ----
    if (kb + 32 < K) {
      const int gr = mBase + aRow;
      if (gr < M) __builtin_prefetch(A + (size_t)gr * K + kb + 32 + aCol, 0, 0);
      __builtin_prefetch(B + (size_t)(kb + 32 + bK0) * 128 + bN0, 0, 0);
    }

    // ---- stage A 128x32 tile (fp32 -> bf16, optional GELU) ----
    {
      const int gr = mBase + aRow;
      v8bf lo, hi;
      if (gr < M) {
        const float4* s4 = (const float4*)(A + (size_t)gr * K + kb + aCol);
        float4 f0 = s4[0], f1 = s4[1], f2 = s4[2], f3 = s4[3];
        if (gelu_in) {
          f0.x = gelu_exact(f0.x); f0.y = gelu_exact(f0.y); f0.z = gelu_exact(f0.z); f0.w = gelu_exact(f0.w);
          f1.x = gelu_exact(f1.x); f1.y = gelu_exact(f1.y); f1.z = gelu_exact(f1.z); f1.w = gelu_exact(f1.w);
          f2.x = gelu_exact(f2.x); f2.y = gelu_exact(f2.y); f2.z = gelu_exact(f2.z); f2.w = gelu_exact(f2.w);
          f3.x = gelu_exact(f3.x); f3.y = gelu_exact(f3.y); f3.z = gelu_exact(f3.z); f3.w = gelu_exact(f3.w);
        }
        lo = v8bf{(bf16_t)f0.x, (bf16_t)f0.y, (bf16_t)f0.z, (bf16_t)f0.w,
                  (bf16_t)f1.x, (bf16_t)f1.y, (bf16_t)f1.z, (bf16_t)f1.w};
        hi = v8bf{(bf16_t)f2.x, (bf16_t)f2.y, (bf16_t)f2.z, (bf16_t)f2.w,
                  (bf16_t)f3.x, (bf16_t)f3.y, (bf16_t)f3.z, (bf16_t)f3.w};
      } else {
        lo = v8bf{}; hi = v8bf{};
      }
      *(v8bf*)&As[aRow][aCol]     = lo;   // 16B aligned (80B pitch)
      *(v8bf*)&As[aRow][aCol + 8] = hi;
    }
    // ---- stage B 32x128 tile transposed via 4x4 register micro-transpose ----
    {
      const float* Bp = B + (size_t)(kb + bK0) * 128 + bN0;
      float4 r0 = *(const float4*)(Bp);
      float4 r1 = *(const float4*)(Bp + 128);
      float4 r2 = *(const float4*)(Bp + 256);
      float4 r3 = *(const float4*)(Bp + 384);
      *(v4bf*)&Bt[bN0 + 0][bK0] = v4bf{(bf16_t)r0.x, (bf16_t)r1.x, (bf16_t)r2.x, (bf16_t)r3.x};
      *(v4bf*)&Bt[bN0 + 1][bK0] = v4bf{(bf16_t)r0.y, (bf16_t)r1.y, (bf16_t)r2.y, (bf16_t)r3.y};
      *(v4bf*)&Bt[bN0 + 2][bK0] = v4bf{(bf16_t)r0.z, (bf16_t)r1.z, (bf16_t)r2.z, (bf16_t)r3.z};
      *(v4bf*)&Bt[bN0 + 3][bK0] = v4bf{(bf16_t)r0.w, (bf16_t)r1.w, (bf16_t)r2.w, (bf16_t)r3.w};
    }
    __syncthreads();

    // ---- load A fragment + all 8 B fragments, then dense WMMA burst ----
    union { v16bf v; v8bf h[2]; } au;
    au.h[0] = *(const v8bf*)&As[mr][kh];
    au.h[1] = *(const v8bf*)&As[mr][kh + 16];

    union { v16bf v; v8bf h[2]; } bu[8];
#pragma unroll
    for (int nt = 0; nt < 8; ++nt) {
      const bf16_t* bp = &Bt[nt * 16 + colb][kh2];
      bu[nt].h[0] = *(const v8bf*)(bp);
      bu[nt].h[1] = *(const v8bf*)(bp + 16);
    }
#pragma unroll
    for (int nt = 0; nt < 8; ++nt) {
      acc[nt] = __builtin_amdgcn_wmma_f32_16x16x32_bf16(
          false, au.v, false, bu[nt].v, (short)0, acc[nt], false, false);
    }
    __syncthreads();
  }

  // ---- epilogue: bias, activation, sigmoid-skip residual ----
  float sk = 0.f, osk = 0.f;
  if (resid) {
    float sv = skip ? skip[0] : 0.f;
    sk  = 1.f / (1.f + expf(-sv));
    osk = 1.f - sk;
  }
  const int rowHi = (lane >> 4) * 8;   // C layout: VGPR j -> M=j (lo lanes), M=j+8 (hi lanes)
  const int col   = lane & 15;
#pragma unroll
  for (int nt = 0; nt < 8; ++nt) {
    const int gc = nt * 16 + col;
#pragma unroll
    for (int j = 0; j < 8; ++j) {
      const int gr = mBase + wv * 16 + rowHi + j;
      if (gr < M) {
        float v = acc[nt][j];
        if (bias) v += bias[gc];
        if (act_out == 1) v = leaky01(v);
        if (resid) v = sk * v + osk * resid[(size_t)gr * 128 + gc];
        C[(size_t)gr * 128 + gc] = v;
      }
    }
  }
}

// ---------------------------------------------------------------------------
// User encoder: thread-per-user scalar chain (tiny K: 4/5 -> 32|32 -> 64 -> 128)
// ---------------------------------------------------------------------------
__global__ void user_encoder(const float* __restrict__ cat4,
                             const float* __restrict__ num5,
                             const float* __restrict__ Wc, const float* __restrict__ bc,
                             const float* __restrict__ Wn, const float* __restrict__ bn,
                             const float* __restrict__ Wp, const float* __restrict__ bp,
                             const float* __restrict__ Wpo, const float* __restrict__ bpo,
                             float* __restrict__ out, int U)
{
  int u = blockIdx.x * blockDim.x + threadIdx.x;
  if (u >= U) return;
  float ci[4], ni[5], h1[64], h2[64];
#pragma unroll
  for (int i = 0; i < 4; ++i) ci[i] = cat4[(size_t)u * 4 + i];
#pragma unroll
  for (int i = 0; i < 5; ++i) ni[i] = num5[(size_t)u * 5 + i];
  for (int j = 0; j < 32; ++j) {
    float s = bc[j];
#pragma unroll
    for (int i = 0; i < 4; ++i) s += ci[i] * Wc[i * 32 + j];
    h1[j] = leaky01(s);
  }
  for (int j = 0; j < 32; ++j) {
    float s = bn[j];
#pragma unroll
    for (int i = 0; i < 5; ++i) s += ni[i] * Wn[i * 32 + j];
    h1[32 + j] = leaky01(s);
  }
  for (int j = 0; j < 64; ++j) {
    float s = bp[j];
    for (int i = 0; i < 64; ++i) s += h1[i] * Wp[i * 64 + j];
    h2[j] = leaky01(s);
  }
  for (int j = 0; j < 128; ++j) {
    float s = bpo[j];
    for (int i = 0; i < 64; ++i) s += h2[i] * Wpo[i * 128 + j];
    out[(size_t)u * 128 + j] = leaky01(s);
  }
}

// ---------------------------------------------------------------------------
// Edge attention kernels (wave-per-edge; lane i owns float4 at offset 4*i, 32*4=128)
// ---------------------------------------------------------------------------
__device__ __forceinline__ void atomicMaxF(float* addr, float val) {
  unsigned int* a = (unsigned int*)addr;
  unsigned int old = *a;
  while (__uint_as_float(old) < val) {
    unsigned int assumed = old;
    old = atomicCAS(a, assumed, __float_as_uint(val));
    if (old == assumed) break;
  }
}

__global__ __launch_bounds__(256)
void edge_score(const int* __restrict__ src, const int* __restrict__ dst, int E,
                const float* __restrict__ q,     // [n_dst,128]
                const float* __restrict__ krel,  // [n_src,128]
                const float* __restrict__ prel,  // scalar
                float inv_sqrt_d,
                float* __restrict__ a, float* __restrict__ m)
{
  int e = blockIdx.x * 8 + (threadIdx.x >> 5);
  int lane = threadIdx.x & 31;
  if (e >= E) return;
  int s = src[e], d = dst[e];
  float4 qv = ((const float4*)(q    + (size_t)d * 128))[lane];
  float4 kv = ((const float4*)(krel + (size_t)s * 128))[lane];
  float p = qv.x * kv.x + qv.y * kv.y + qv.z * kv.z + qv.w * kv.w;
#pragma unroll
  for (int off = 16; off > 0; off >>= 1) p += __shfl_xor(p, off, 32);
  p *= prel[0] * inv_sqrt_d;
  if (lane == 0) { a[e] = p; atomicMaxF(&m[d], p); }
}

__global__ void edge_exp(const int* __restrict__ dst, int E,
                         float* __restrict__ a,      // in: score, out: exp(score-max)
                         const float* __restrict__ m,
                         float* __restrict__ z)
{
  int e = blockIdx.x * blockDim.x + threadIdx.x;
  if (e >= E) return;
  int d = dst[e];
  float ev = expf(a[e] - m[d]);
  a[e] = ev;
  atomicAdd(&z[d], ev);
}

__global__ __launch_bounds__(256)
void edge_aggregate(const int* __restrict__ src, const int* __restrict__ dst, int E,
                    const float* __restrict__ a,    // exp values
                    const float* __restrict__ z,
                    const float* __restrict__ vrel, // [n_src,128]
                    float* __restrict__ agg)        // [n_dst,128]
{
  int e = blockIdx.x * 8 + (threadIdx.x >> 5);
  int lane = threadIdx.x & 31;
  if (e >= E) return;
  int s = src[e], d = dst[e];
  float coeff = a[e] / (z[d] + 1e-16f);
  float4 vv = ((const float4*)(vrel + (size_t)s * 128))[lane];
  float* out = agg + (size_t)d * 128 + lane * 4;
  atomicAdd(out + 0, vv.x * coeff);
  atomicAdd(out + 1, vv.y * coeff);
  atomicAdd(out + 2, vv.z * coeff);
  atomicAdd(out + 3, vv.w * coeff);
}

// ---------------------------------------------------------------------------
// Classifier head layer 2: [M,128] @ [128,2] + bc2 (thread-per-row)
// ---------------------------------------------------------------------------
__global__ void classify2(const float* __restrict__ z1,
                          const float* __restrict__ Wc2, const float* __restrict__ bc2,
                          float* __restrict__ out, int M)
{
  int r = blockIdx.x * blockDim.x + threadIdx.x;
  if (r >= M) return;
  float s0 = bc2[0], s1 = bc2[1];
  const float* zr = z1 + (size_t)r * 128;
  for (int i = 0; i < 128; ++i) {
    float zi = zr[i];
    s0 += zi * Wc2[i * 2 + 0];
    s1 += zi * Wc2[i * 2 + 1];
  }
  out[(size_t)r * 2 + 0] = s0;
  out[(size_t)r * 2 + 1] = s1;
}

// ---------------------------------------------------------------------------
// host-side orchestration
// ---------------------------------------------------------------------------
static inline unsigned div_up(size_t a, size_t b) { return (unsigned)((a + b - 1) / b); }

extern "C" void kernel_launch(void* const* d_in, const int* in_sizes, int n_in,
                              void* d_out, int out_size, void* d_ws, size_t ws_size,
                              hipStream_t stream) {
  // Input order: setup_inputs() insertion order; `params` dict expands to its
  // leaves in insertion order at its slot.
  // 0 node_type | 1 cat_prop | 2 num_prop | 3 des | 4 text
  // 5 Wc 6 bc 7 Wn 8 bn 9 Wp 10 bp 11 Wpo 12 bpo 13 Wt 14 bt
  // 15 Wc1 16 bc1 17 Wc2 18 bc2
  // 19 Wk_u 20 bk_u 21 Wq_u 22 bq_u 23 Wv_u 24 bv_u 25 Wa_u 26 ba_u 27 skip_u
  // 28 Wk_t 29 bk_t 30 Wq_t 31 bq_t 32 Wv_t 33 bv_t 34 Wa_t 35 ba_t 36 skip_t
  // 37 arel_follow 38 mrel_follow 39 prel_follow
  // 40 arel_friend 41 mrel_friend 42 prel_friend
  // 43 arel_post   44 mrel_post   45 prel_post
  // 46 follow_edges 47 friend_edges 48 post_edges
  const float* cat_prop = (const float*)d_in[1];
  const float* num_prop = (const float*)d_in[2];
  const float* text     = (const float*)d_in[4];
  const float* P[46];
  for (int i = 5; i < 46 && i < n_in; ++i) P[i] = (const float*)d_in[i];
  const int* follow  = (const int*)d_in[46];
  const int* friendE = (const int*)d_in[47];
  const int* postE   = (const int*)d_in[48];

  const int U = in_sizes[1] / 4;      // 50000
  const int T = in_sizes[4] / 768;    // 100000
  const int E = in_sizes[46] / 2;     // 200000
  const float inv_sqrt_d = 0.08838834764831845f;   // 1/sqrt(128)

  float* ws = (float*)d_ws;
  const size_t UD = (size_t)U * 128, TD = (size_t)T * 128;
  float* x_user  = ws;               // [U,128]
  float* x_tweet = x_user  + UD;     // [T,128]
  float* q_user  = x_tweet + TD;     // [U,128]
  float* q_tweet = q_user  + UD;     // [T,128]
  float* k_user  = q_tweet + TD;     // [U,128]
  float* v_user  = k_user  + UD;     // [U,128]
  float* k_rel   = v_user  + UD;     // [U,128] (reused per edge type)
  float* v_rel   = k_rel   + UD;     // [U,128]
  float* agg_u   = v_rel   + UD;     // [U,128]
  float* agg_t   = agg_u   + UD;     // [T,128]
  float* e_buf   = agg_t   + TD;     // [E]
  float* m_buf   = e_buf   + E;      // [max(U,T)]
  float* z_buf   = m_buf   + T;      // [max(U,T)]
  // reuse: hu -> k_user, ht -> q_tweet, z1_user -> q_user, z1_tweet -> agg_t
  float* hu  = k_user;
  float* ht  = q_tweet;
  float* z1u = q_user;
  float* z1t = agg_t;

  float* out = (float*)d_out;

  // init accumulators (workspace is poisoned)
  fill_f32<<<div_up(UD, 256), 256, 0, stream>>>(agg_u, UD, 0.f);
  fill_f32<<<div_up(TD, 256), 256, 0, stream>>>(agg_t, TD, 0.f);

  // node encoders
  user_encoder<<<div_up((size_t)U, 256), 256, 0, stream>>>(
      cat_prop, num_prop, P[5], P[6], P[7], P[8], P[9], P[10], P[11], P[12], x_user, U);
  gemm128_wmma_bf16<<<div_up((size_t)T, 128), 256, 0, stream>>>(
      text, T, 768, P[13], P[14], x_tweet, /*leaky*/1, 0, nullptr, nullptr);

  // HGT K/Q/V (tweet K/V are unused by any edge type -> skipped)
  gemm128_wmma_bf16<<<div_up((size_t)U, 128), 256, 0, stream>>>(
      x_user, U, 128, P[21], P[22], q_user, 0, 0, nullptr, nullptr);
  gemm128_wmma_bf16<<<div_up((size_t)U, 128), 256, 0, stream>>>(
      x_user, U, 128, P[19], P[20], k_user, 0, 0, nullptr, nullptr);
  gemm128_wmma_bf16<<<div_up((size_t)U, 128), 256, 0, stream>>>(
      x_user, U, 128, P[23], P[24], v_user, 0, 0, nullptr, nullptr);
  gemm128_wmma_bf16<<<div_up((size_t)T, 128), 256, 0, stream>>>(
      x_tweet, T, 128, P[30], P[31], q_tweet, 0, 0, nullptr, nullptr);

  // edge types: {arel, mrel, prel, edges, q_dst, n_dst, agg}
  struct ET { int ai, mi, pi; const int* ed; const float* q; int nd; float* agg; };
  ET ets[3] = {
      {37, 38, 39, follow,  q_user,  U, agg_u},
      {40, 41, 42, friendE, q_user,  U, agg_u},
      {43, 44, 45, postE,   q_tweet, T, agg_t},
  };
  for (int t = 0; t < 3; ++t) {
    const ET& et = ets[t];
    const int* src = et.ed;
    const int* dst = et.ed + E;
    gemm128_wmma_bf16<<<div_up((size_t)U, 128), 256, 0, stream>>>(
        k_user, U, 128, P[et.ai], nullptr, k_rel, 0, 0, nullptr, nullptr);
    gemm128_wmma_bf16<<<div_up((size_t)U, 128), 256, 0, stream>>>(
        v_user, U, 128, P[et.mi], nullptr, v_rel, 0, 0, nullptr, nullptr);
    fill_f32<<<div_up((size_t)et.nd, 256), 256, 0, stream>>>(m_buf, et.nd, -INFINITY);
    fill_f32<<<div_up((size_t)et.nd, 256), 256, 0, stream>>>(z_buf, et.nd, 0.f);
    edge_score<<<div_up((size_t)E, 8), 256, 0, stream>>>(
        src, dst, E, et.q, k_rel, P[et.pi], inv_sqrt_d, e_buf, m_buf);
    edge_exp<<<div_up((size_t)E, 256), 256, 0, stream>>>(dst, E, e_buf, m_buf, z_buf);
    edge_aggregate<<<div_up((size_t)E, 8), 256, 0, stream>>>(
        src, dst, E, e_buf, z_buf, v_rel, et.agg);
  }

  // node_out: h = sigmoid(skip)*(gelu(agg)@Wa + ba) + (1-sigmoid(skip))*x
  gemm128_wmma_bf16<<<div_up((size_t)U, 128), 256, 0, stream>>>(
      agg_u, U, 128, P[25], P[26], hu, 0, /*gelu_in*/1, x_user, P[27]);
  gemm128_wmma_bf16<<<div_up((size_t)T, 128), 256, 0, stream>>>(
      agg_t, T, 128, P[34], P[35], ht, 0, /*gelu_in*/1, x_tweet, P[36]);

  // classifier layer 1 (leaky) then tiny layer 2
  gemm128_wmma_bf16<<<div_up((size_t)U, 128), 256, 0, stream>>>(
      hu, U, 128, P[15], P[16], z1u, 1, 0, nullptr, nullptr);
  gemm128_wmma_bf16<<<div_up((size_t)T, 128), 256, 0, stream>>>(
      ht, T, 128, P[15], P[16], z1t, 1, 0, nullptr, nullptr);
  classify2<<<div_up((size_t)U, 256), 256, 0, stream>>>(z1u, P[17], P[18], out, U);
  classify2<<<div_up((size_t)T, 256), 256, 0, stream>>>(z1t, P[17], P[18], out + (size_t)U * 2, T);

  (void)out_size; (void)ws_size; (void)n_in;
}
